// EncoderLayer_64982855188497
// MI455X (gfx1250) — compile-verified, matrix operand order checked by reading, not run on hardware
//
#include <hip/hip_runtime.h>
#include <hip/hip_bf16.h>
#include <math.h>

// ---------------------------------------------------------------------------
// Types
// ---------------------------------------------------------------------------
typedef __attribute__((ext_vector_type(16))) __bf16        v16bf;
typedef __attribute__((ext_vector_type(8)))  __bf16        v8bf;
typedef __attribute__((ext_vector_type(8)))  float         v8f;
typedef __attribute__((ext_vector_type(8)))  unsigned int  v8u;
typedef __attribute__((ext_vector_type(4)))  unsigned int  u32x4;
typedef __attribute__((ext_vector_type(8)))  int           i32x8;
typedef __attribute__((ext_vector_type(4)))  int           i32x4;

#define EPS 1e-5f

__device__ __forceinline__ __bf16 f2bf(float f) {
  unsigned int u = __builtin_bit_cast(unsigned int, f);
  u += 0x7FFFu + ((u >> 16) & 1u);          // round-to-nearest-even
  unsigned short h = (unsigned short)(u >> 16);
  return __builtin_bit_cast(__bf16, h);
}

__device__ __forceinline__ float gelu_exact(float v) {
  return 0.5f * v * (1.0f + erff(v * 0.70710678118654752f));
}

// ---------------------------------------------------------------------------
// f32 -> bf16 elementwise convert (weights)
// ---------------------------------------------------------------------------
__global__ void conv_bf16_kernel(const float* __restrict__ in,
                                 __bf16* __restrict__ out, int n) {
  int i = blockIdx.x * blockDim.x + threadIdx.x;
  if (i < n) out[i] = f2bf(in[i]);
}

// ---------------------------------------------------------------------------
// LayerNorm: f32 [rows x D] -> bf16, one 256-thread block per row (D = 1024)
// ---------------------------------------------------------------------------
__global__ void ln_kernel(const float* __restrict__ x,
                          const float* __restrict__ g,
                          const float* __restrict__ b,
                          __bf16* __restrict__ out, int D) {
  __shared__ float red[8];
  const int row = blockIdx.x, tid = threadIdx.x;
  const int lane = tid & 31, wid = tid >> 5;
  const float* xr = x + (size_t)row * D;

  float xv[4];
  float s = 0.f;
#pragma unroll
  for (int i = 0; i < 4; ++i) { xv[i] = xr[tid + 256 * i]; s += xv[i]; }
#pragma unroll
  for (int o = 16; o > 0; o >>= 1) s += __shfl_xor(s, o, 32);
  if (lane == 0) red[wid] = s;
  __syncthreads();
  if (wid == 0) {
    float t = (lane < 8) ? red[lane] : 0.f;
#pragma unroll
    for (int o = 4; o > 0; o >>= 1) t += __shfl_xor(t, o, 32);
    if (lane == 0) red[0] = t;
  }
  __syncthreads();
  const float mean = red[0] / (float)D;
  __syncthreads();

  float vs = 0.f;
#pragma unroll
  for (int i = 0; i < 4; ++i) { float d = xv[i] - mean; vs += d * d; }
#pragma unroll
  for (int o = 16; o > 0; o >>= 1) vs += __shfl_xor(vs, o, 32);
  if (lane == 0) red[wid] = vs;
  __syncthreads();
  if (wid == 0) {
    float t = (lane < 8) ? red[lane] : 0.f;
#pragma unroll
    for (int o = 4; o > 0; o >>= 1) t += __shfl_xor(t, o, 32);
    if (lane == 0) red[0] = t;
  }
  __syncthreads();
  const float inv = rsqrtf(red[0] / (float)D + EPS);

#pragma unroll
  for (int i = 0; i < 4; ++i) {
    int c = tid + 256 * i;
    out[(size_t)row * D + c] = f2bf((xv[i] - mean) * inv * g[c] + b[c]);
  }
}

// ---------------------------------------------------------------------------
// Row softmax: f32 [rows x L] -> bf16 P, one 256-thread block per row (L=2048)
// ---------------------------------------------------------------------------
__global__ void softmax_kernel(const float* __restrict__ S,
                               __bf16* __restrict__ P, int L) {
  __shared__ float red[8];
  const int row = blockIdx.x, tid = threadIdx.x;
  const int lane = tid & 31, wid = tid >> 5;
  const float* sr = S + (size_t)row * L;

  float sv[8];
  float m = -3.0e38f;
#pragma unroll
  for (int i = 0; i < 8; ++i) { sv[i] = sr[tid + 256 * i]; m = fmaxf(m, sv[i]); }
#pragma unroll
  for (int o = 16; o > 0; o >>= 1) m = fmaxf(m, __shfl_xor(m, o, 32));
  if (lane == 0) red[wid] = m;
  __syncthreads();
  if (wid == 0) {
    float t = (lane < 8) ? red[lane] : -3.0e38f;
#pragma unroll
    for (int o = 4; o > 0; o >>= 1) t = fmaxf(t, __shfl_xor(t, o, 32));
    if (lane == 0) red[0] = t;
  }
  __syncthreads();
  const float rowmax = red[0];
  __syncthreads();

  float s = 0.f;
#pragma unroll
  for (int i = 0; i < 8; ++i) { sv[i] = __expf(sv[i] - rowmax); s += sv[i]; }
#pragma unroll
  for (int o = 16; o > 0; o >>= 1) s += __shfl_xor(s, o, 32);
  if (lane == 0) red[wid] = s;
  __syncthreads();
  if (wid == 0) {
    float t = (lane < 8) ? red[lane] : 0.f;
#pragma unroll
    for (int o = 4; o > 0; o >>= 1) t += __shfl_xor(t, o, 32);
    if (lane == 0) red[0] = t;
  }
  __syncthreads();
  const float inv = 1.0f / red[0];

#pragma unroll
  for (int i = 0; i < 8; ++i)
    P[(size_t)row * L + tid + 256 * i] = f2bf(sv[i] * inv);
}

// ---------------------------------------------------------------------------
// Split QKV: qkv bf16 [Ntok x 3D] -> q[B,H,L,dk], kT[B,H,dk,L], v[B,H,L,dk]
// ---------------------------------------------------------------------------
__global__ void split_qkv_kernel(const __bf16* __restrict__ qkv,
                                 __bf16* __restrict__ q,
                                 __bf16* __restrict__ kT,
                                 __bf16* __restrict__ v, int total) {
  const int Dd = 1024, Ld = 2048, Hd = 16;
  int idx = blockIdx.x * blockDim.x + threadIdx.x;
  if (idx >= total) return;
  int n = idx / (3 * Dd);
  int c = idx - n * (3 * Dd);
  int bb = n / Ld;
  int l  = n - bb * Ld;
  int which = c / Dd;
  int dD = c - which * Dd;
  int h  = dD >> 6;
  int dd = dD & 63;
  __bf16 val = qkv[idx];
  size_t hb = (size_t)(bb * Hd + h);
  if (which == 0)      q [(hb * Ld + l) * 64 + dd] = val;
  else if (which == 1) kT[(hb * 64 + dd) * Ld + l] = val;
  else                 v [(hb * Ld + l) * 64 + dd] = val;
}

// ---------------------------------------------------------------------------
// Generic WMMA bf16 GEMM: C[M,N] = A[M,K] * B[K,N]  (row-major, f32 acc)
//   BM=128, BN=WN*32, BK=32; 256 threads (8 waves, 4x2); 32x(WN*16) per wave.
//   A tile staged by the Tensor Data Mover (tensor_load_to_lds, TENSORcnt);
//   B tile staged transposed so bf16 B-fragments assemble from b32 K-pairs.
//   mode 0: out bf16 (+bias if non-null)
//   mode 1: out bf16, +bias, exact GELU
//   mode 2: out f32 * scale           (attention scores)
//   mode 3: out f32, +bias, +residual (proj / ff2 epilogues)
// ---------------------------------------------------------------------------
template <int WN>
__global__ __launch_bounds__(256) void gemm_bf16_kernel(
    int M, int N, int K,
    const __bf16* __restrict__ A, int lda,
    const __bf16* __restrict__ B, int ldb,
    const float* __restrict__ bias,
    void* __restrict__ outp, int ldc,
    const float* __restrict__ res, int ldr,
    float scale, int mode) {
  constexpr int BN = WN * 32;
  __shared__ __align__(32) __bf16 As[128 * 32];  // [row][k]
  __shared__ __align__(32) __bf16 Bs[BN * 32];   // [col][k]  (transposed tile)

  const int tid  = threadIdx.x;
  const int lane = tid & 31;
  const int wid  = tid >> 5;
  const int wm   = wid & 3;   // 0..3 -> 32-row slices
  const int wn   = wid >> 2;  // 0..1 -> (WN*16)-col slices
  const int mblk = blockIdx.y * 128;
  const int nblk = blockIdx.x * BN;

  v8f acc[2][WN] = {};

  const unsigned int* As32 = (const unsigned int*)As;
  const unsigned int* Bs32 = (const unsigned int*)Bs;
  const int hi4 = (lane >> 4) << 2;  // K half-select per ISA bf16 A/B layout

  for (int k0 = 0; k0 < K; k0 += 32) {
    __syncthreads();  // all waves done reading the previous tile

#if defined(__HIP_DEVICE_COMPILE__)
    if (wid == 0) {  // one TDM descriptor per block: A tile 128x32, stride lda
      unsigned long long ga =
          (unsigned long long)(const void*)(A + (size_t)mblk * lda + k0);
      unsigned ldsa = (unsigned)(unsigned long long)(const void*)&As[0];
      u32x4 g0;
      g0[0] = 1u;                                    // count=1 (valid D#)
      g0[1] = ldsa;                                  // lds_addr
      g0[2] = (unsigned)ga;                          // global_addr[31:0]
      g0[3] = (unsigned)(ga >> 32) | 0x80000000u;    // addr[56:32] | type=2
      i32x8 g1;
      g1[0] = 0x00010000;          // data_size = 2 bytes
      g1[1] = (int)(32u << 16);    // tensor_dim0 = 32
      g1[2] = (int)(128u << 16);   // tensor_dim1 = 128
      g1[3] = (int)(32u << 16);    // tile_dim0 = 32
      g1[4] = 128;                 // tile_dim1 = 128, tile_dim2 = 0
      g1[5] = lda;                 // tensor_dim0_stride (elements)
      g1[6] = 0;
      g1[7] = 0;
      i32x4 z4 = {};
#if __has_include(<hip/amd_detail/amd_gfx1250_TDM.h>)
      i32x8 z8 = {};
      __builtin_amdgcn_tensor_load_to_lds(g0, g1, z4, z4, z8, 0);
#else
      __builtin_amdgcn_tensor_load_to_lds(g0, g1, z4, z4, 0);
#endif
    }
#else
    {  // host-pass type-check fallback (never executed on device)
      int r  = tid >> 1;
      int c0 = (tid & 1) * 16;
      v16bf av = *(const v16bf*)(A + (size_t)(mblk + r) * lda + k0 + c0);
      *(v16bf*)&As[r * 32 + c0] = av;
    }
#endif

    {  // stage B tile transposed: 32 x BN -> Bs[col][k]
      int kk = tid >> 3;
      int c0 = (tid & 7) * (BN / 8);
      if constexpr (WN == 4) {
        v16bf bv = *(const v16bf*)(B + (size_t)(k0 + kk) * ldb + nblk + c0);
#pragma unroll
        for (int i = 0; i < 16; ++i) Bs[(c0 + i) * 32 + kk] = bv[i];
      } else {
        v8bf bv = *(const v8bf*)(B + (size_t)(k0 + kk) * ldb + nblk + c0);
#pragma unroll
        for (int i = 0; i < 8; ++i) Bs[(c0 + i) * 32 + kk] = bv[i];
      }
    }

    if (k0 + 32 < K)  // CDNA5 global_prefetch_b8 of next B tile
      __builtin_prefetch(B + (size_t)(k0 + 32 + (tid >> 3)) * ldb + nblk, 0, 1);

#if defined(__HIP_DEVICE_COMPILE__)
    if (wid == 0) __builtin_amdgcn_s_wait_tensorcnt(0);  // A tile landed
#endif
    __syncthreads();  // publish A (TDM) + B (ds stores) to all waves

    v16bf af[2];
    v16bf bfg[WN];
#pragma unroll
    for (int f = 0; f < 2; ++f) {
      const int r = wm * 32 + f * 16 + (lane & 15);
      v8u ta;
#pragma unroll
      for (int vv = 0; vv < 8; ++vv) {
        int kk = ((vv < 4) ? vv : vv + 4) + hi4;  // dword index of K-pair
        ta[vv] = As32[r * 16 + kk];
      }
      af[f] = __builtin_bit_cast(v16bf, ta);
    }
#pragma unroll
    for (int f = 0; f < WN; ++f) {
      const int c = wn * (WN * 16) + f * 16 + (lane & 15);
      v8u tb;
#pragma unroll
      for (int vv = 0; vv < 8; ++vv) {
        int kk = ((vv < 4) ? vv : vv + 4) + hi4;
        tb[vv] = Bs32[c * 16 + kk];
      }
      bfg[f] = __builtin_bit_cast(v16bf, tb);
    }

#pragma unroll
    for (int fm = 0; fm < 2; ++fm)
#pragma unroll
      for (int fn = 0; fn < WN; ++fn)
        acc[fm][fn] = __builtin_amdgcn_wmma_f32_16x16x32_bf16(
            false, af[fm], false, bfg[fn], (short)0, acc[fm][fn], false, false);
  }

  // Epilogue. C layout: VGPR i -> row (i + 8*(lane/16)), col (lane%16).
  float*  outF = (float*)outp;
  __bf16* outH = (__bf16*)outp;
#pragma unroll
  for (int fm = 0; fm < 2; ++fm)
#pragma unroll
    for (int fn = 0; fn < WN; ++fn)
#pragma unroll
      for (int i = 0; i < 8; ++i) {
        int row = mblk + wm * 32 + fm * 16 + i + ((lane >> 4) << 3);
        int col = nblk + wn * (WN * 16) + fn * 16 + (lane & 15);
        float v = acc[fm][fn][i];
        size_t o = (size_t)row * ldc + col;
        if (mode == 0) {
          if (bias) v += bias[col];
          outH[o] = f2bf(v);
        } else if (mode == 1) {
          outH[o] = f2bf(gelu_exact(v + bias[col]));
        } else if (mode == 2) {
          outF[o] = v * scale;
        } else {
          outF[o] = v + bias[col] + res[(size_t)row * ldr + col];
        }
      }
}

// ---------------------------------------------------------------------------
// Host orchestration
// ---------------------------------------------------------------------------
extern "C" void kernel_launch(void* const* d_in, const int* in_sizes, int n_in,
                              void* d_out, int out_size, void* d_ws, size_t ws_size,
                              hipStream_t stream) {
  (void)in_sizes; (void)n_in; (void)out_size; (void)ws_size;
  const int D = 1024, H = 16, FF = 4096, Bb = 2, L = 2048;
  const int Ntok = Bb * L;

  const float* x      = (const float*)d_in[0];
  const float* ln1_g  = (const float*)d_in[1];
  const float* ln1_b  = (const float*)d_in[2];
  const float* qkv_w  = (const float*)d_in[3];
  const float* proj_w = (const float*)d_in[4];
  const float* proj_b = (const float*)d_in[5];
  const float* ln2_g  = (const float*)d_in[6];
  const float* ln2_b  = (const float*)d_in[7];
  const float* w1     = (const float*)d_in[8];
  const float* b1     = (const float*)d_in[9];
  const float* w2     = (const float*)d_in[10];
  const float* b2     = (const float*)d_in[11];

  char* ws = (char*)d_ws;
  size_t off = 0;
  auto alloc = [&](size_t bytes) {
    size_t o = off;
    off = (off + bytes + 255) & ~(size_t)255;
    return o;
  };
  __bf16* Wqkv = (__bf16*)(ws + alloc((size_t)D * 3 * D * 2));
  __bf16* Wp   = (__bf16*)(ws + alloc((size_t)D * D * 2));
  __bf16* W1   = (__bf16*)(ws + alloc((size_t)D * FF * 2));
  __bf16* W2   = (__bf16*)(ws + alloc((size_t)FF * D * 2));
  __bf16* xn   = (__bf16*)(ws + alloc((size_t)Ntok * D * 2));
  __bf16* qkv  = (__bf16*)(ws + alloc((size_t)Ntok * 3 * D * 2));
  __bf16* q    = (__bf16*)(ws + alloc((size_t)Ntok * D * 2));
  __bf16* kT   = (__bf16*)(ws + alloc((size_t)Ntok * D * 2));
  __bf16* v    = (__bf16*)(ws + alloc((size_t)Ntok * D * 2));
  __bf16* ctx  = (__bf16*)(ws + alloc((size_t)Ntok * D * 2));
  float*  scr  = (float*) (ws + alloc((size_t)L * L * 4));
  __bf16* Pb   = (__bf16*)(ws + alloc((size_t)L * L * 2));
  float*  x1   = (float*) (ws + alloc((size_t)Ntok * D * 4));
  __bf16* h2   = (__bf16*)(ws + alloc((size_t)Ntok * D * 2));
  __bf16* h1   = (__bf16*)(ws + alloc((size_t)Ntok * FF * 2));

  auto conv = [&](const float* src, __bf16* dst, int n) {
    conv_bf16_kernel<<<(n + 255) / 256, 256, 0, stream>>>(src, dst, n);
  };
  auto gemm128 = [&](int M, int N, int K, const __bf16* A, int lda,
                     const __bf16* Bm, int ldb, const float* bias, void* out,
                     int ldc, const float* res, int ldr, float scale, int mode) {
    dim3 g(N / 128, M / 128);
    gemm_bf16_kernel<4><<<g, 256, 0, stream>>>(M, N, K, A, lda, Bm, ldb, bias,
                                               out, ldc, res, ldr, scale, mode);
  };
  auto gemm64 = [&](int M, int N, int K, const __bf16* A, int lda,
                    const __bf16* Bm, int ldb, const float* bias, void* out,
                    int ldc, const float* res, int ldr, float scale, int mode) {
    dim3 g(N / 64, M / 128);
    gemm_bf16_kernel<2><<<g, 256, 0, stream>>>(M, N, K, A, lda, Bm, ldb, bias,
                                               out, ldc, res, ldr, scale, mode);
  };

  // 1) weights -> bf16
  conv(qkv_w,  Wqkv, D * 3 * D);
  conv(proj_w, Wp,   D * D);
  conv(w1,     W1,   D * FF);
  conv(w2,     W2,   FF * D);

  // 2) LN1
  ln_kernel<<<Ntok, 256, 0, stream>>>(x, ln1_g, ln1_b, xn, D);

  // 3) QKV projection: [Ntok,D] x [D,3D]
  gemm128(Ntok, 3 * D, D, xn, D, Wqkv, 3 * D, nullptr, qkv, 3 * D,
          nullptr, 0, 1.0f, 0);

  // 4) split into per-head q / k^T / v
  {
    int total = Ntok * 3 * D;
    split_qkv_kernel<<<(total + 255) / 256, 256, 0, stream>>>(qkv, q, kT, v, total);
  }

  // 5) attention per (b,h): scores -> softmax -> P*V (stream-ordered)
  for (int bh = 0; bh < Bb * H; ++bh) {
    int bb = bh / H, h = bh % H;
    const __bf16* qh  = q  + (size_t)bh * L * 64;
    const __bf16* kTh = kT + (size_t)bh * 64 * L;
    const __bf16* vh  = v  + (size_t)bh * L * 64;
    gemm128(L, L, 64, qh, 64, kTh, L, nullptr, scr, L, nullptr, 0, 0.125f, 2);
    softmax_kernel<<<L, 256, 0, stream>>>(scr, Pb, L);
    __bf16* ctxh = ctx + (size_t)bb * L * D + h * 64;
    gemm64(L, 64, L, Pb, L, vh, 64, nullptr, ctxh, D, nullptr, 0, 1.0f, 0);
  }

  // 6) output projection + residual: x1 = x + ctx @ Wp + proj_b
  gemm128(Ntok, D, D, ctx, D, Wp, D, proj_b, x1, D, x, D, 1.0f, 3);

  // 7) LN2
  ln_kernel<<<Ntok, 256, 0, stream>>>(x1, ln2_g, ln2_b, h2, D);

  // 8) FF1 + exact GELU
  gemm128(Ntok, FF, D, h2, D, W1, FF, b1, h1, FF, nullptr, 0, 1.0f, 1);

  // 9) FF2 + residual -> d_out
  gemm128(Ntok, D, FF, h1, FF, W2, D, b2, (float*)d_out, D, x1, D, 1.0f, 3);
}
